// NerfSDModule_79869211837017
// MI455X (gfx1250) — matrix operand order repaired and checked by reading, not compiled, not run on hardware
//
#include <hip/hip_runtime.h>
#include <hip/hip_bf16.h>

typedef __attribute__((ext_vector_type(16))) __bf16 v16bf;
typedef __attribute__((ext_vector_type(8)))  float  v8f;

#define HW_    1024
#define DD_    32
#define CC_    128
#define KPAD_  352
#define INDIM_ 326

static __device__ __forceinline__ __bf16 tobf(float f) {
  union { float f; unsigned u; } in; in.f = f;
  unsigned u = in.u;
  unsigned r = u + 0x7FFFu + ((u >> 16) & 1u);     // round-to-nearest-even
  unsigned short hs = (unsigned short)(r >> 16);
  union { unsigned short s; __bf16 b; } out; out.s = hs;
  return out.b;
}

// ---------------- weight conversion: f32 [K][N] -> bf16 [N][Kpad] ----------------
__global__ void conv_weights(const float* __restrict__ W1, const float* __restrict__ W2,
                             __bf16* __restrict__ w1t, __bf16* __restrict__ w2t) {
  int i = blockIdx.x * blockDim.x + threadIdx.x;
  if (i < CC_ * KPAD_) {                  // W1^T padded: n = i/352, k = i%352
    int n = i / KPAD_, k = i - n * KPAD_;
    w1t[i] = tobf(k < INDIM_ ? W1[k * CC_ + n] : 0.f);
  }
  if (i < CC_ * CC_) {                    // W2^T: n = i>>7, k = i&127
    int n = i >> 7, k = i & 127;
    w2t[i] = tobf(W2[k * CC_ + n]);
  }
}

// ---------------- fused forward ----------------
__global__ __launch_bounds__(128)
void nerf_fused(const float* __restrict__ xref, const float* __restrict__ mask,
                const float* __restrict__ rp,   const float* __restrict__ rays,
                const float* __restrict__ R,    const float* __restrict__ T,
                const float* __restrict__ focal,
                const float* __restrict__ b1v,  const float* __restrict__ b2v,
                const float* __restrict__ Wv,   const float* __restrict__ bv,
                const float* __restrict__ Wd,
                const __bf16* __restrict__ W1t, const __bf16* __restrict__ W2t,
                float* __restrict__ out)
{
  __shared__ __bf16 lds_A[16 * KPAD_];     // A tile (plane_in rows), bf16
  __shared__ __bf16 lds_h[16 * CC_];       // silu(GEMM1) rows, bf16
  __shared__ float  lds_pf[4 * 16 * CC_];  // pf_final per view
  __shared__ float  lds_coef[16][4];
  __shared__ int    lds_base[16][4];
  __shared__ float  lds_geo[16][12];       // pv[0:3], plucker[3:9], otg[9:12]
  __shared__ float  lds_lacc[16];
  __shared__ float  lds_logit[4][16];
  __shared__ float  lds_attn[4][16];
  __shared__ float  lds_l0[16];
  __shared__ float  lds_red[16];

  const int tid  = threadIdx.x;
  const int lane = tid & 31;
  const int wave = tid >> 5;
  const int r    = tid & 15;          // row within tile
  const int p    = tid >> 4;          // slot 0..7
  const int m    = blockIdx.x * 16 + r;
  const int h    = m >> 5;
  const int d    = m & 31;

  // ---- F0: view-independent logit part (xyz_grid enc from view 0) ----
  float pts0 = 0.f, pts1 = 0.f, pts2 = 0.f;
  if (tid < 16) {
    pts0 = rp[m * 3 + 0]; pts1 = rp[m * 3 + 1]; pts2 = rp[m * 3 + 2];
    float pv0[3];
    #pragma unroll
    for (int j = 0; j < 3; ++j)
      pv0[j] = pts0 * R[j] + pts1 * R[3 + j] + pts2 * R[6 + j] + T[j];
    float l0 = bv[0];
    #pragma unroll
    for (int k = 0; k < 3; ++k) {
      float fr = 1.f;
      for (int f = 0; f < 16; ++f) {
        float a = pv0[k] * fr;
        l0 += sinf(a) * Wv[128 + k * 32 + f] + cosf(a) * Wv[128 + k * 32 + 16 + f];
        fr *= 2.f;
      }
    }
    l0 += pv0[0] * Wv[224] + pv0[1] * Wv[225] + pv0[2] * Wv[226];
    lds_l0[r] = l0;
  }

  for (int v = 0; v < 4; ++v) {
    const int iv = v + 1;
    __syncthreads();

    // ---- G: per-row scalar geometry (threads 0..15) ----
    if (tid < 16) {
      const float* Rv = R + iv * 9;
      const float* Tv = T + iv * 3;
      __bf16* Arow = lds_A + r * KPAD_;
      float pv[3];
      #pragma unroll
      for (int j = 0; j < 3; ++j) {
        pv[j] = pts0 * Rv[j] + pts1 * Rv[3 + j] + pts2 * Rv[6 + j] + Tv[j];
        lds_geo[r][j] = pv[j];
        Arow[224 + j] = tobf(pv[j]);
      }
      // NDC grid -> bilinear coeffs (zero-pad OOB, mask folded in)
      float fz = focal[iv];
      float gx = -fz * pv[0] / pv[2];
      float gy = -fz * pv[1] / pv[2];
      if (isnan(gx)) gx = 0.f;
      if (isnan(gy)) gy = 0.f;
      gx = fminf(fmaxf(gx, -1.2f), 1.2f);
      gy = fminf(fmaxf(gy, -1.2f), 1.2f);
      float x = (gx + 1.f) * 15.5f, y = (gy + 1.f) * 15.5f;
      float x0f = floorf(x), y0f = floorf(y);
      float wx1 = x - x0f, wx0 = 1.f - wx1;
      float wy1 = y - y0f, wy0 = 1.f - wy1;
      int ix0 = (int)x0f, iy0 = (int)y0f;
      int cx[4] = {ix0, ix0 + 1, ix0, ix0 + 1};
      int cy[4] = {iy0, iy0, iy0 + 1, iy0 + 1};
      float cw[4] = {wx0 * wy0, wx1 * wy0, wx0 * wy1, wx1 * wy1};
      #pragma unroll
      for (int c4 = 0; c4 < 4; ++c4) {
        int xi = cx[c4], yi = cy[c4];
        float valid = (xi >= 0 && xi <= 31 && yi >= 0 && yi <= 31) ? 1.f : 0.f;
        int xc = min(max(xi, 0), 31), yc = min(max(yi, 0), 31);
        lds_coef[r][c4] = cw[c4] * valid * mask[v * 1024 + yc * 32 + xc];
        lds_base[r][c4] = (yc * 32 + xc) * CC_;
      }
      // cam_iv: o/d in view frame, plucker
      float ot[3], dt[3];
      #pragma unroll
      for (int k = 0; k < 3; ++k) { ot[k] = rays[h * 6 + k]; dt[k] = rays[h * 6 + 3 + k]; }
      float oiv[3], dvv[3];
      #pragma unroll
      for (int j = 0; j < 3; ++j) {
        oiv[j] = ot[0] * Rv[j] + ot[1] * Rv[3 + j] + ot[2] * Rv[6 + j] + Tv[j];
        dvv[j] = dt[0] * Rv[j] + dt[1] * Rv[3 + j] + dt[2] * Rv[6 + j];
      }
      float nrm = sqrtf(dvv[0]*dvv[0] + dvv[1]*dvv[1] + dvv[2]*dvv[2]) + 1e-8f;
      float dn0 = dvv[0] / nrm, dn1 = dvv[1] / nrm, dn2 = dvv[2] / nrm;
      lds_geo[r][3] = dn0; lds_geo[r][4] = dn1; lds_geo[r][5] = dn2;
      lds_geo[r][6] = oiv[1] * dn2 - oiv[2] * dn1;
      lds_geo[r][7] = oiv[2] * dn0 - oiv[0] * dn2;
      lds_geo[r][8] = oiv[0] * dn1 - oiv[1] * dn0;
      Arow[323] = tobf(dvv[0]); Arow[324] = tobf(dvv[1]); Arow[325] = tobf(dvv[2]);
      for (int k = INDIM_; k < KPAD_; ++k) Arow[k] = tobf(0.f);
      // o_tg
      float og[3], otg[3];
      #pragma unroll
      for (int k = 0; k < 3; ++k) og[k] = rays[(iv * HW_ + h) * 6 + k];
      #pragma unroll
      for (int j = 0; j < 3; ++j) {
        otg[j] = og[0] * R[j] + og[1] * R[3 + j] + og[2] * R[6 + j] + T[j];
        lds_geo[r][9 + j] = otg[j];
      }
      lds_lacc[r] = lds_l0[r] + otg[0] * Wv[227] + otg[1] * Wv[228] + otg[2] * Wv[229];
    }
    __syncthreads();

    // ---- E: encodings + bilinear plane features, all 128 threads ----
    {
      __bf16* Arow = lds_A + r * KPAD_;
      if (p < 3) {
        // xyz_iv_enc dim p: [128 + p*32 .. +32)
        float xv = lds_geo[r][p];
        float fr = 1.f;
        for (int f = 0; f < 16; ++f) {
          float a = xv * fr;
          Arow[128 + p * 32 + f]      = tobf(sinf(a));
          Arow[128 + p * 32 + 16 + f] = tobf(cosf(a));
          fr *= 2.f;
        }
      } else if (p < 6) {
        // plucker enc dims 2*(p-3), 2*(p-3)+1: [227 + k*16 .. +16)
        int k0 = (p - 3) * 2;
        #pragma unroll
        for (int kk = 0; kk < 2; ++kk) {
          int k = k0 + kk;
          float xv = lds_geo[r][3 + k];
          float fr = 1.f;
          for (int f = 0; f < 8; ++f) {
            float a = xv * fr;
            Arow[227 + k * 16 + f]     = tobf(sinf(a));
            Arow[227 + k * 16 + 8 + f] = tobf(cosf(a));
            fr *= 2.f;
          }
        }
      } else {
        // o_tg posenc logit contribution: p==6 -> dims 0,1 ; p==7 -> dim 2
        int kb = (p == 6) ? 0 : 2;
        int ke = (p == 6) ? 2 : 3;
        float lp = 0.f;
        for (int k = kb; k < ke; ++k) {
          float xv = lds_geo[r][9 + k];
          float fr = 1.f;
          for (int f = 0; f < 16; ++f) {
            float a = xv * fr;
            lp += sinf(a) * Wv[230 + k * 32 + f] + cosf(a) * Wv[230 + k * 32 + 16 + f];
            fr *= 2.f;
          }
        }
        atomicAdd(&lds_lacc[r], lp);
      }
      // bilinear plane features: channels [p*16, p*16+16)
      const float* xv = xref + v * HW_ * CC_;
      float c0 = lds_coef[r][0], c1 = lds_coef[r][1], c2 = lds_coef[r][2], c3 = lds_coef[r][3];
      int   e0 = lds_base[r][0], e1 = lds_base[r][1], e2 = lds_base[r][2], e3 = lds_base[r][3];
      float lpart = 0.f;
      for (int cc = p * 16; cc < p * 16 + 16; ++cc) {
        float val = c0 * xv[e0 + cc] + c1 * xv[e1 + cc] + c2 * xv[e2 + cc] + c3 * xv[e3 + cc];
        Arow[cc] = tobf(val);
        lpart += val * Wv[cc];
      }
      atomicAdd(&lds_lacc[r], lpart);
    }
    __syncthreads();
    if (tid < 16) lds_logit[v][r] = lds_lacc[r];

    // ---- GEMM1: h = silu(A @ W1 + b1), bf16 WMMA 16x16x32, K=352 ----
    {
      const int Mr = lane & 15, hi = lane >> 4, nb = wave * 32;
      const int N0 = nb + (lane & 15), N1 = N0 + 16;
      v8f acc0 = {0.f,0.f,0.f,0.f,0.f,0.f,0.f,0.f};
      v8f acc1 = acc0;
      #pragma unroll
      for (int kb = 0; kb < KPAD_; kb += 32) {
        v16bf a, bA, bB;
        const __bf16* Ar = lds_A + Mr * KPAD_;
        #pragma unroll
        for (int s = 0; s < 4; ++s) {
          int k  = kb + hi * 8 + 2 * s;
          int k2 = kb + 16 + hi * 8 + 2 * s;
          a[2*s]   = Ar[k];  a[2*s+1] = Ar[k + 1];
          a[2*s+8] = Ar[k2]; a[2*s+9] = Ar[k2 + 1];
        }
        const __bf16* cA = W1t + N0 * KPAD_;
        const __bf16* cB = W1t + N1 * KPAD_;
        #pragma unroll
        for (int s = 0; s < 4; ++s) {
          int k  = kb + hi * 8 + 2 * s;
          int k2 = kb + 16 + hi * 8 + 2 * s;
          bA[2*s] = cA[k]; bA[2*s+1] = cA[k + 1]; bA[2*s+8] = cA[k2]; bA[2*s+9] = cA[k2 + 1];
          bB[2*s] = cB[k]; bB[2*s+1] = cB[k + 1]; bB[2*s+8] = cB[k2]; bB[2*s+9] = cB[k2 + 1];
        }
        acc0 = __builtin_amdgcn_wmma_f32_16x16x32_bf16(false, a, false, bA, (short)0, acc0, false, false);
        acc1 = __builtin_amdgcn_wmma_f32_16x16x32_bf16(false, a, false, bB, (short)0, acc1, false, false);
      }
      float bn0 = b1v[N0], bn1 = b1v[N1];
      #pragma unroll
      for (int g = 0; g < 8; ++g) {
        int row = g + 8 * hi;
        float x0 = acc0[g] + bn0, x1 = acc1[g] + bn1;
        lds_h[row * CC_ + N0] = tobf(x0 / (1.f + expf(-x0)));
        lds_h[row * CC_ + N1] = tobf(x1 / (1.f + expf(-x1)));
      }
    }
    __syncthreads();

    // ---- GEMM2: pf = h @ W2 + b2, K=128 ----
    {
      const int Mr = lane & 15, hi = lane >> 4, nb = wave * 32;
      const int N0 = nb + (lane & 15), N1 = N0 + 16;
      v8f acc0 = {0.f,0.f,0.f,0.f,0.f,0.f,0.f,0.f};
      v8f acc1 = acc0;
      #pragma unroll
      for (int kb = 0; kb < CC_; kb += 32) {
        v16bf a, bA, bB;
        const __bf16* Ar = lds_h + Mr * CC_;
        #pragma unroll
        for (int s = 0; s < 4; ++s) {
          int k  = kb + hi * 8 + 2 * s;
          int k2 = kb + 16 + hi * 8 + 2 * s;
          a[2*s]   = Ar[k];  a[2*s+1] = Ar[k + 1];
          a[2*s+8] = Ar[k2]; a[2*s+9] = Ar[k2 + 1];
        }
        const __bf16* cA = W2t + N0 * CC_;
        const __bf16* cB = W2t + N1 * CC_;
        #pragma unroll
        for (int s = 0; s < 4; ++s) {
          int k  = kb + hi * 8 + 2 * s;
          int k2 = kb + 16 + hi * 8 + 2 * s;
          bA[2*s] = cA[k]; bA[2*s+1] = cA[k + 1]; bA[2*s+8] = cA[k2]; bA[2*s+9] = cA[k2 + 1];
          bB[2*s] = cB[k]; bB[2*s+1] = cB[k + 1]; bB[2*s+8] = cB[k2]; bB[2*s+9] = cB[k2 + 1];
        }
        acc0 = __builtin_amdgcn_wmma_f32_16x16x32_bf16(false, a, false, bA, (short)0, acc0, false, false);
        acc1 = __builtin_amdgcn_wmma_f32_16x16x32_bf16(false, a, false, bB, (short)0, acc1, false, false);
      }
      float bn0 = b2v[N0], bn1 = b2v[N1];
      #pragma unroll
      for (int g = 0; g < 8; ++g) {
        int row = g + 8 * hi;
        lds_pf[(v * 16 + row) * CC_ + N0] = acc0[g] + bn0;
        lds_pf[(v * 16 + row) * CC_ + N1] = acc1[g] + bn1;
      }
    }
  } // view loop

  __syncthreads();
  // ---- softmax over views ----
  if (tid < 16) {
    float l0 = lds_logit[0][r], l1 = lds_logit[1][r], l2 = lds_logit[2][r], l3 = lds_logit[3][r];
    float mx = fmaxf(fmaxf(l0, l1), fmaxf(l2, l3));
    float e0 = expf(l0 - mx), e1 = expf(l1 - mx), e2 = expf(l2 - mx), e3 = expf(l3 - mx);
    float s = e0 + e1 + e2 + e3;
    lds_attn[0][r] = e0 / s; lds_attn[1][r] = e1 / s;
    lds_attn[2][r] = e2 / s; lds_attn[3][r] = e3 / s;
    lds_red[r] = 0.f;
  }
  __syncthreads();
  // ---- fuse + write fused, partial dot with Wd ----
  {
    float a0 = lds_attn[0][r], a1 = lds_attn[1][r], a2 = lds_attn[2][r], a3 = lds_attn[3][r];
    float part = 0.f;
    for (int cc = p * 16; cc < p * 16 + 16; ++cc) {
      float f = a0 * lds_pf[(0 * 16 + r) * CC_ + cc] + a1 * lds_pf[(1 * 16 + r) * CC_ + cc]
              + a2 * lds_pf[(2 * 16 + r) * CC_ + cc] + a3 * lds_pf[(3 * 16 + r) * CC_ + cc];
      out[m * 129 + cc] = f;
      part += f * Wd[cc];
    }
    atomicAdd(&lds_red[r], part);
  }
  __syncthreads();
  if (tid < 16) {
    out[m * 129 + 128] = lds_red[r];
    const int OUT0 = HW_ * DD_ * 129;
    #pragma unroll
    for (int v = 0; v < 4; ++v)
      out[OUT0 + (v * HW_ + h) * DD_ + d] = lds_attn[v][r];
  }
}

extern "C" void kernel_launch(void* const* d_in, const int* in_sizes, int n_in,
                              void* d_out, int out_size, void* d_ws, size_t ws_size,
                              hipStream_t stream) {
  const float* xref  = (const float*)d_in[0];
  const float* maskr = (const float*)d_in[1];
  const float* rp    = (const float*)d_in[2];
  const float* rays  = (const float*)d_in[3];
  const float* R     = (const float*)d_in[4];
  const float* T     = (const float*)d_in[5];
  const float* focal = (const float*)d_in[6];
  const float* W1    = (const float*)d_in[7];
  const float* b1    = (const float*)d_in[8];
  const float* W2    = (const float*)d_in[9];
  const float* b2    = (const float*)d_in[10];
  const float* Wv    = (const float*)d_in[11];
  const float* bv    = (const float*)d_in[12];
  const float* Wd    = (const float*)d_in[13];
  float* out = (float*)d_out;

  __bf16* w1t = (__bf16*)d_ws;                 // 128*352 bf16
  __bf16* w2t = w1t + CC_ * KPAD_;             // 128*128 bf16

  conv_weights<<<(CC_ * KPAD_ + 255) / 256, 256, 0, stream>>>(W1, W2, w1t, w2t);

  const int M = HW_ * DD_;
  nerf_fused<<<M / 16, 128, 0, stream>>>(xref, maskr, rp, rays, R, T, focal,
                                         b1, b2, Wv, bv, Wd, w1t, w2t, out);
  (void)in_sizes; (void)n_in; (void)out_size; (void)ws_size;
}
